// MultiHeadSelfAttention_52939766891187
// MI455X (gfx1250) — matrix operand order repaired and checked by reading, not compiled, not run on hardware
//
#include <hip/hip_runtime.h>
#include <hip/hip_bf16.h>
#include <math.h>

// ---------------------------------------------------------------------------
// MultiHeadSelfAttention block for MI455X (gfx1250, wave32, WMMA 16x16x32 f16)
// B=2, C=256, H=W=64 -> N=4096, heads=8, hd=32, groups=8
//
// All WMMA operands are stored FRAGMENT-PACKED:
//  B-tile (32K x 16N): 32 lanes x 32B, lane L (n=L&15, hs=L>>4): same k order as A-tile
//  A-tile (16M x 32K): 32 lanes x 32B, lane L (m=L&15, hs=L>>4):
//        half j<8 : k = hs*8 + j ;  half j>=8 : k = 16 + hs*8 + (j-8)
// so every fragment load is one contiguous 32B (2x b128) per lane.
// ---------------------------------------------------------------------------

typedef __attribute__((ext_vector_type(16))) _Float16 v16h;
typedef __attribute__((ext_vector_type(8)))  float    v8f;

#define BATCH   2
#define CDIM    256
#define NTOK    4096
#define NHEADS  8
#define HDIM    32
#define NGROUPS 8
#define GCH     (CDIM / NGROUPS)        // 32 channels per group
#define GELEMS  (GCH * NTOK)            // 131072 elements per (b, group)
#define TILEH   512                     // halves per packed 32x16 / 16x32 tile
#define BHQ     (NTOK / 16 * TILEH)     // halves per (b,head) packed q/k/v: 131072

static __device__ __forceinline__ v8f wmma_f16(v16h a, v16h b, v8f c) {
    return __builtin_amdgcn_wmma_f32_16x16x32_f16(
        false, a, false, b, (short)0, c, false, false);
}

// A fragment from a row-major 16x32 f16 tile (used for the weight matrices;
// per-lane pairs are contiguous so these merge into global_load_b128).
static __device__ __forceinline__ v16h load_a_frag_rm(const _Float16* A, int lda, int lane) {
    int m  = lane & 15;
    int hs = lane >> 4;
    const _Float16* row = A + (size_t)m * lda;
    v16h a;
#pragma unroll
    for (int i = 0; i < 4; ++i) {
        int k0 = hs * 8 + 2 * i;
        a[2 * i]     = row[k0];
        a[2 * i + 1] = row[k0 + 1];
        int k1 = 16 + hs * 8 + 2 * i;
        a[2 * (i + 4)]     = row[k1];
        a[2 * (i + 4) + 1] = row[k1 + 1];
    }
    return a;
}

// Packed-A address helpers: value (m, k) of a 16x32 tile lives at
// lane_t = m + 16*((k&15)>>3), half jj = ((k>>4)<<3) + (k&7).
static __device__ __forceinline__ int apack_off(int m, int k) {
    int lane_t = m + 16 * ((k & 15) >> 3);
    int jj     = ((k >> 4) << 3) + (k & 7);
    return lane_t * 16 + jj;
}
// Packed-B: value (k, n) of a 32x16 tile lives at lane' = n + 16*((k&15)>>3),
// half j = ((k>>4)<<3) + (k&7).
static __device__ __forceinline__ int bpack_off(int k, int n) {
    return (n + 16 * ((k & 15) >> 3)) * 16 + (((k >> 4) << 3) + (k & 7));
}

// ---------------------------------------------------------------------------
// f32 -> f16 cast
__global__ void cvt_f16_kernel(const float* __restrict__ src,
                               _Float16* __restrict__ dst, int n) {
    int i = blockIdx.x * 256 + threadIdx.x;
    if (i < n) dst[i] = (_Float16)src[i];
}

// ---------------------------------------------------------------------------
// GroupNorm statistics: one block per (b, group)
__global__ void gn_stats_kernel(const float* __restrict__ x,
                                float* __restrict__ mean,
                                float* __restrict__ rstd) {
    __shared__ float sh[512];
    int bg = blockIdx.x;                  // 0..15
    int b  = bg >> 3;
    int g  = bg & 7;
    const float* base = x + ((size_t)b * CDIM + (size_t)g * GCH) * NTOK;
    float s = 0.f, ss = 0.f;
    for (int i = threadIdx.x; i < GELEMS; i += 256) {
        float v = base[i];
        s  += v;
        ss += v * v;
    }
    sh[threadIdx.x]       = s;
    sh[256 + threadIdx.x] = ss;
    __syncthreads();
    for (int st = 128; st > 0; st >>= 1) {
        if (threadIdx.x < st) {
            sh[threadIdx.x]       += sh[threadIdx.x + st];
            sh[256 + threadIdx.x] += sh[256 + threadIdx.x + st];
        }
        __syncthreads();
    }
    if (threadIdx.x == 0) {
        float m   = sh[0] * (1.0f / GELEMS);
        float var = sh[256] * (1.0f / GELEMS) - m * m;
        mean[bg] = m;
        rstd[bg] = rsqrtf(var + 1e-5f);
    }
}

// Normalize + affine + cast, written B-fragment-packed:
// hnPk[b][nt=n>>4][kt=c>>5][512]
__global__ void gn_apply_kernel(const float* __restrict__ x,
                                const float* __restrict__ mean,
                                const float* __restrict__ rstd,
                                const float* __restrict__ w,
                                const float* __restrict__ bias,
                                _Float16* __restrict__ hnPk) {
    size_t i = (size_t)blockIdx.x * 256 + threadIdx.x;   // < B*C*N
    int n = (int)(i & 4095);
    int c = (int)((i >> 12) & 255);
    int b = (int)(i >> 20);
    int bg = b * 8 + (c >> 5);
    float v = (x[i] - mean[bg]) * rstd[bg] * w[c] + bias[c];
    size_t idx = (((size_t)b * 256 + (n >> 4)) * 8 + (c >> 5)) * TILEH
               + bpack_off(c & 31, n & 15);
    hnPk[idx] = (_Float16)v;
}

// ---------------------------------------------------------------------------
// QKV GEMM: qkv[o][n] = sum_c wq[o][c] * hn[b][c][n] + bqkv[o]
// M=768, N=4096, K=256 per batch.  One wave -> one 16x16 tile.
// B fragments: one contiguous 32B global load per lane (packed layout).
// Outputs written fragment-packed for the attention kernel:
//   q -> qPk[bh][qt][A-packed 16qn x 32d]
//   k -> kPk[bh][nt][B-packed 32d x 16kn]
//   v -> vPk[bh][kc][d-tile][B-packed 32kn x 16d]
__global__ void qkv_gemm_kernel(const _Float16* __restrict__ wq16,
                                const float* __restrict__ bqkv,
                                const _Float16* __restrict__ hnPk,
                                _Float16* __restrict__ qPk,
                                _Float16* __restrict__ kPk,
                                _Float16* __restrict__ vPk) {
    int wid  = threadIdx.x >> 5;
    int lane = threadIdx.x & 31;
    int tile = blockIdx.x * 4 + wid;          // 2 * 48 * 256 tiles
    int b    = tile / (48 * 256);
    int rem  = tile % (48 * 256);
    int mt   = rem >> 8;
    int nt   = rem & 255;

    const _Float16* A  = wq16 + (size_t)mt * 16 * CDIM;
    const _Float16* Bp = hnPk + (((size_t)b * 256 + nt) * 8) * TILEH + lane * 16;

    v8f acc = {};
#pragma unroll
    for (int kt = 0; kt < 8; ++kt) {
        v16h a  = load_a_frag_rm(A + kt * 32, CDIM, lane);
        v16h bf = *(const v16h*)(Bp + (size_t)kt * TILEH);
        acc = wmma_f16(a, bf, acc);
    }

    int hs = lane >> 4;
    int nl = lane & 15;
    int n  = nt * 16 + nl;
#pragma unroll
    for (int r = 0; r < 8; ++r) {
        int o = mt * 16 + r + 8 * hs;
        float val = acc[r] + bqkv[o];
        int sect = o >> 8;                    // 0:q 1:k 2:v
        int c = o & 255;
        int h = c >> 5, d = c & 31;
        size_t bh = (size_t)b * NHEADS + h;
        _Float16 hv = (_Float16)val;
        if (sect == 0) {
            // A-packed Q tile (rows = token n, cols k = d)
            qPk[bh * BHQ + (size_t)(n >> 4) * TILEH + apack_off(n & 15, d)] = hv;
        } else if (sect == 1) {
            // B-packed K tile (k = d, n = token)
            kPk[bh * BHQ + (size_t)(n >> 4) * TILEH + bpack_off(d, n & 15)] = hv;
        } else {
            // B-packed V tile (k = token within 32-chunk, n = d within 16-tile)
            vPk[bh * BHQ + (size_t)(n >> 5) * (2 * TILEH) + (size_t)(d >> 4) * TILEH
                + bpack_off(n & 31, d & 15)] = hv;
        }
    }
}

// ---------------------------------------------------------------------------
// Flash attention. One block = 4 waves sharing one (b, head); each wave owns
// a 16-row q tile.  Per 32-wide k chunk the block stages the 2KB packed K and
// V chunks into LDS with gfx1250 async DMA (global_load_async_to_lds_b128 +
// s_wait_asynccnt), then each wave: S = Q*K^T (2 WMMAs) -> online softmax
// (shuffle reductions in 16-lane halves matching the C-fragment rows) -> P
// scattered into a per-wave A-packed LDS tile -> O += P*V (2 WMMAs).
__global__ void attn_kernel(const _Float16* __restrict__ qPk,
                            const _Float16* __restrict__ kPk,
                            const _Float16* __restrict__ vPk,
                            _Float16* __restrict__ oPk) {
    __shared__ __align__(32) _Float16 kChunk[2 * TILEH];   // 2 B-tiles (2KB)
    __shared__ __align__(32) _Float16 vChunk[2 * TILEH];   // 2 B-tiles (2KB)
    __shared__ __align__(32) _Float16 pbuf[4][TILEH];      // A-packed P per wave

    int wid  = threadIdx.x >> 5;
    int lane = threadIdx.x & 31;
    int tid  = threadIdx.x;
    int bh   = blockIdx.x >> 6;              // 16 (b,head) pairs
    int qt   = (blockIdx.x & 63) * 4 + wid;  // 256 q tiles per (b,head)

    const _Float16* Qb = qPk + (size_t)bh * BHQ + (size_t)qt * TILEH;
    const _Float16* Kb = kPk + (size_t)bh * BHQ;
    const _Float16* Vb = vPk + (size_t)bh * BHQ;
    _Float16* pb = pbuf[wid];

    v16h aq = *(const v16h*)(Qb + lane * 16);   // packed A fragment of Q

    const float scale = 0.17677669529663687f;   // 1/sqrt(32)
    float mrow[8], lrow[8], al[8];
    v8f o0 = {}, o1 = {};
#pragma unroll
    for (int r = 0; r < 8; ++r) { mrow[r] = -INFINITY; lrow[r] = 0.f; }

    int hs  = lane >> 4;
    int col = lane & 15;
    int poff0 = apack_off(0, col);        // P scatter offsets (row part added later)
    int poff1 = apack_off(0, 16 + col);
    int plt   = 16 * ((col >> 3) << 4);   // lane_t*16 contribution without m

    for (int kb = 0; kb < NTOK; kb += 32) {
        // ---- async DMA stage of packed K and V chunks (16B per thread) ----
        {
            const _Float16* kg = Kb + (size_t)(kb >> 4) * TILEH + tid * 8;
            const _Float16* vg = Vb + (size_t)(kb >> 5) * (2 * TILEH) + tid * 8;
            unsigned kl = (unsigned)(uintptr_t)&kChunk[tid * 8];
            unsigned vl = (unsigned)(uintptr_t)&vChunk[tid * 8];
            unsigned long long kga = (unsigned long long)(uintptr_t)kg;
            unsigned long long vga = (unsigned long long)(uintptr_t)vg;
            asm volatile("global_load_async_to_lds_b128 %0, %1, off"
                         :: "v"(kl), "v"(kga) : "memory");
            asm volatile("global_load_async_to_lds_b128 %0, %1, off"
                         :: "v"(vl), "v"(vga) : "memory");
            asm volatile("s_wait_asynccnt 0x0" ::: "memory");
        }
        __syncthreads();

        // ---- S = Q * K^T over the 32-wide chunk (two 16-col tiles) ----
        v16h bk0 = *(const v16h*)&kChunk[lane * 16];
        v16h bk1 = *(const v16h*)&kChunk[TILEH + lane * 16];
        v8f z = {};
        v8f s0 = wmma_f16(aq, bk0, z);
        v8f s1 = wmma_f16(aq, bk1, z);

#pragma unroll
        for (int r = 0; r < 8; ++r) {
            float a0 = s0[r] * scale;
            float a1 = s1[r] * scale;
            float cm = fmaxf(a0, a1);
            cm = fmaxf(cm, __shfl_xor(cm, 1, 16));
            cm = fmaxf(cm, __shfl_xor(cm, 2, 16));
            cm = fmaxf(cm, __shfl_xor(cm, 4, 16));
            cm = fmaxf(cm, __shfl_xor(cm, 8, 16));
            float mn    = fmaxf(mrow[r], cm);
            float alpha = __expf(mrow[r] - mn);
            float p0    = __expf(a0 - mn);
            float p1    = __expf(a1 - mn);
            float rs = p0 + p1;
            rs += __shfl_xor(rs, 1, 16);
            rs += __shfl_xor(rs, 2, 16);
            rs += __shfl_xor(rs, 4, 16);
            rs += __shfl_xor(rs, 8, 16);
            lrow[r] = lrow[r] * alpha + rs;
            mrow[r] = mn;
            al[r]   = alpha;
            // scatter P into the per-wave A-packed tile
            int m = r + 8 * hs;
            pb[m * 16 + poff0] = (_Float16)p0;
            pb[m * 16 + poff1] = (_Float16)p1;
        }
        (void)plt;
#pragma unroll
        for (int r = 0; r < 8; ++r) { o0[r] *= al[r]; o1[r] *= al[r]; }

        // ---- reload P as packed A fragment, accumulate O += P * V ----
        v16h ap  = *(const v16h*)&pb[lane * 16];
        v16h bv0 = *(const v16h*)&vChunk[lane * 16];
        v16h bv1 = *(const v16h*)&vChunk[TILEH + lane * 16];
        o0 = wmma_f16(ap, bv0, o0);
        o1 = wmma_f16(ap, bv1, o1);

        __syncthreads();                     // before next chunk overwrites LDS
    }

    // normalize and store O fragment-packed for the projection GEMM:
    // oPk[b][nt=qn>>4][kt=c>>5==h][B-packed 32c x 16qn]
    int b = bh >> 3, h = bh & 7;
    _Float16* ob = oPk + (((size_t)b * 256 + qt) * 8 + h) * TILEH;
#pragma unroll
    for (int r = 0; r < 8; ++r) {
        int m = r + 8 * hs;
        float inv = 1.0f / lrow[r];
        ob[bpack_off(col,      m)] = (_Float16)(o0[r] * inv);
        ob[bpack_off(16 + col, m)] = (_Float16)(o1[r] * inv);
    }
}

// ---------------------------------------------------------------------------
// Output projection: out[b][o][n] = sum_c wp[o][c] * oPk[b][c][n] + bp[o]
__global__ void proj_gemm_kernel(const _Float16* __restrict__ wp16,
                                 const float* __restrict__ bp,
                                 const _Float16* __restrict__ oPk,
                                 float* __restrict__ out) {
    int wid  = threadIdx.x >> 5;
    int lane = threadIdx.x & 31;
    int tile = blockIdx.x * 4 + wid;          // 2 * 16 * 256 tiles
    int b    = tile / (16 * 256);
    int rem  = tile % (16 * 256);
    int mt   = rem >> 8;
    int nt   = rem & 255;

    const _Float16* A  = wp16 + (size_t)mt * 16 * CDIM;
    const _Float16* Bp = oPk + (((size_t)b * 256 + nt) * 8) * TILEH + lane * 16;

    v8f acc = {};
#pragma unroll
    for (int kt = 0; kt < 8; ++kt) {
        v16h a  = load_a_frag_rm(A + kt * 32, CDIM, lane);
        v16h bf = *(const v16h*)(Bp + (size_t)kt * TILEH);
        acc = wmma_f16(a, bf, acc);
    }

    int hs = lane >> 4;
    int n  = nt * 16 + (lane & 15);
#pragma unroll
    for (int r = 0; r < 8; ++r) {
        int o = mt * 16 + r + 8 * hs;
        out[((size_t)b * CDIM + o) * NTOK + n] = acc[r] + bp[o];
    }
}

// ---------------------------------------------------------------------------
extern "C" void kernel_launch(void* const* d_in, const int* in_sizes, int n_in,
                              void* d_out, int out_size, void* d_ws, size_t ws_size,
                              hipStream_t stream) {
    (void)in_sizes; (void)n_in; (void)out_size; (void)ws_size;

    const float* x    = (const float*)d_in[0];
    const float* gw   = (const float*)d_in[1];
    const float* gb   = (const float*)d_in[2];
    const float* wqkv = (const float*)d_in[3];
    const float* bqkv = (const float*)d_in[4];
    const float* wprj = (const float*)d_in[5];
    const float* bprj = (const float*)d_in[6];
    float* out = (float*)d_out;

    // workspace carve-up (256B aligned)
    char* ws = (char*)d_ws;
    size_t off = 0;
    auto alloc = [&](size_t bytes) {
        char* p = ws + off;
        off += (bytes + 255) & ~(size_t)255;
        return p;
    };
    float*    mean = (float*)alloc(16 * sizeof(float));
    float*    rstd = (float*)alloc(16 * sizeof(float));
    _Float16* hnPk = (_Float16*)alloc((size_t)BATCH * CDIM * NTOK * 2);
    _Float16* wq16 = (_Float16*)alloc((size_t)3 * CDIM * CDIM * 2);
    _Float16* wp16 = (_Float16*)alloc((size_t)CDIM * CDIM * 2);
    _Float16* qPk  = (_Float16*)alloc((size_t)BATCH * NHEADS * BHQ * 2);
    _Float16* kPk  = (_Float16*)alloc((size_t)BATCH * NHEADS * BHQ * 2);
    _Float16* vPk  = (_Float16*)alloc((size_t)BATCH * NHEADS * BHQ * 2);
    _Float16* oPk  = (_Float16*)alloc((size_t)BATCH * CDIM * NTOK * 2);

    const int nWq = 3 * CDIM * CDIM;   // 196608
    const int nWp = CDIM * CDIM;       // 65536

    cvt_f16_kernel<<<(nWq + 255) / 256, 256, 0, stream>>>(wqkv, wq16, nWq);
    cvt_f16_kernel<<<(nWp + 255) / 256, 256, 0, stream>>>(wprj, wp16, nWp);

    gn_stats_kernel<<<BATCH * NGROUPS, 256, 0, stream>>>(x, mean, rstd);
    gn_apply_kernel<<<(BATCH * CDIM * NTOK) / 256, 256, 0, stream>>>(
        x, mean, rstd, gw, gb, hnPk);

    // 2 batches * 48 Mtiles * 256 Ntiles / 4 waves-per-block
    qkv_gemm_kernel<<<(BATCH * 48 * 256) / 4, 128, 0, stream>>>(
        wq16, bqkv, hnPk, qPk, kPk, vPk);

    // 16 (b,h) * 64 q-tile groups (4 waves each)
    attn_kernel<<<(BATCH * NHEADS * 256) / 4, 128, 0, stream>>>(qPk, kPk, vPk, oPk);

    // 2 batches * 16 Mtiles * 256 Ntiles / 4 waves-per-block
    proj_gemm_kernel<<<(BATCH * 16 * 256) / 4, 128, 0, stream>>>(
        wp16, bprj, oPk, out);
}
